// MoEFFNLayer_17970143167046
// MI455X (gfx1250) — matrix operand order repaired and checked by reading, not compiled, not run on hardware
//
#include <hip/hip_runtime.h>
#include <hip/hip_bf16.h>

// ---------------------------------------------------------------------------
// MoE FFN (top-2 of 8 experts, SwiGLU) for gfx1250 (MI455X).
//  Pipeline:
//   0) cvt         : one-shot fp32 -> bf16 weight caches in scratch
//   1) init        : zero counters
//   2) routing     : gate logits -> softmax -> top2 -> counts + block partials
//   3) aux/offsets : deterministic aux loss + expert prefix offsets
//   4) scatter     : per-expert gathered row lists (int atomics only)
//   5) expert      : fused [16-row tile] x (gate_up GEMM -> SiLU*up -> down GEMM)
//                    all matmuls via v_wmma_f32_16x16x32_bf16, bf16 weights
//   6) combine     : out[t] = Y[2t] + Y[2t+1]  (deterministic order)
// ---------------------------------------------------------------------------

#define HID   1024
#define INTER 2048
#define NEXP  8
#define NTOK  8192            // 4 * 2048
#define NROW  (2 * NTOK)      // gathered rows (top-2)
#define BM    16              // rows per expert tile
#define MAXTILES 512          // NTOK / BM  (worst case: one expert gets all)
#define FC    128             // inter columns per fused chunk
#define XS    1040            // LDS stride for x tile (bf16 elems)
#define AS    136             // LDS stride for act tile (bf16 elems)

typedef __attribute__((ext_vector_type(8)))  float  v8f;
typedef __attribute__((ext_vector_type(4)))  __bf16 v4bf;
typedef __attribute__((ext_vector_type(8)))  __bf16 v8bf;
typedef __attribute__((ext_vector_type(16))) __bf16 v16bf;

__device__ __forceinline__ __bf16 f2bf(float f) {
    union { float f; unsigned u; } v; v.f = f;
    unsigned r = v.u + 0x7FFFu + ((v.u >> 16) & 1u);   // round-to-nearest-even
    union { unsigned short s; __bf16 b; } o; o.s = (unsigned short)(r >> 16);
    return o.b;
}

// A-fragment (16x32 bf16): lane m=lane&15, hi=lane>>4 holds
// K = [hi*16, hi*16+16) contiguous.  p = row base + kbase (bf16 elems).
__device__ __forceinline__ v16bf loadA16(const __bf16* p, int hi) {
    v8bf a0 = *(const v8bf*)(p + hi * 16);
    v8bf a1 = *(const v8bf*)(p + hi * 16 + 8);
    return __builtin_shufflevector(a0, a1, 0,1,2,3,4,5,6,7,8,9,10,11,12,13,14,15);
}

// B-fragment (32x16 bf16): lane n=lane&15 is the N column, hi selects K half
// [hi*16, hi*16+16).  p points at this lane's 16 contiguous bf16 weights.
__device__ __forceinline__ v16bf loadB16(const __bf16* __restrict__ p) {
    v8bf b0 = *(const v8bf*)(p);
    v8bf b1 = *(const v8bf*)(p + 8);
    return __builtin_shufflevector(b0, b1, 0,1,2,3,4,5,6,7,8,9,10,11,12,13,14,15);
}

__device__ __forceinline__ v8f wmma_bf16(v16bf a, v16bf b, v8f c) {
    return __builtin_amdgcn_wmma_f32_16x16x32_bf16(false, a, false, b,
                                                   (short)0, c, false, false);
}

// ---------------------------------------------------------------------------
// One-shot fp32 -> bf16 weight cache (bandwidth-bound, ~300MB total traffic).
__global__ __launch_bounds__(256) void cvt_kernel(const float* __restrict__ src,
                                                  __bf16* __restrict__ dst,
                                                  unsigned n8)
{
    unsigned i = blockIdx.x * 256 + threadIdx.x;
    if (i >= n8) return;
    const float4* q = (const float4*)(src + (size_t)i * 8);
    float4 a = q[0], b = q[1];
    v8bf o;
    o[0]=f2bf(a.x); o[1]=f2bf(a.y); o[2]=f2bf(a.z); o[3]=f2bf(a.w);
    o[4]=f2bf(b.x); o[5]=f2bf(b.y); o[6]=f2bf(b.z); o[7]=f2bf(b.w);
    *(v8bf*)(dst + (size_t)i * 8) = o;
}

__global__ void init_kernel(int* cnts) {           // zero counts[8] + cursors[8]
    int i = threadIdx.x;
    if (i < 16) cnts[i] = 0;
}

// One wave32 per token; 8 waves / block.
__global__ __launch_bounds__(256) void routing_kernel(
    const float* __restrict__ x, const float* __restrict__ gw,
    int* __restrict__ topIdx, float* __restrict__ topW,
    int* __restrict__ counts, float* __restrict__ blockPart)
{
    __shared__ float simp[8][NEXP];
    int tid = threadIdx.x, lane = tid & 31, wv = tid >> 5;
    int t = blockIdx.x * 8 + wv;

    float acc[NEXP];
#pragma unroll
    for (int e = 0; e < NEXP; ++e) acc[e] = 0.f;
    for (int j = 0; j < 32; ++j) {
        int h = lane + j * 32;
        float xv = x[(size_t)t * HID + h];
#pragma unroll
        for (int e = 0; e < NEXP; ++e) acc[e] += xv * gw[e * HID + h];
    }
#pragma unroll
    for (int e = 0; e < NEXP; ++e) {
        float s = acc[e];
        for (int o = 16; o; o >>= 1) s += __shfl_xor(s, o, 32);
        acc[e] = s;
    }
    float mx = acc[0];
#pragma unroll
    for (int e = 1; e < NEXP; ++e) mx = fmaxf(mx, acc[e]);
    float p[NEXP], sum = 0.f;
#pragma unroll
    for (int e = 0; e < NEXP; ++e) { p[e] = __expf(acc[e] - mx); sum += p[e]; }
    float inv = __builtin_amdgcn_rcpf(sum);
#pragma unroll
    for (int e = 0; e < NEXP; ++e) p[e] *= inv;

    int i1 = 0; float m1 = p[0];
#pragma unroll
    for (int e = 1; e < NEXP; ++e) if (p[e] > m1) { m1 = p[e]; i1 = e; }
    int i2 = -1; float m2 = -1.f;
#pragma unroll
    for (int e = 0; e < NEXP; ++e) if (e != i1 && p[e] > m2) { m2 = p[e]; i2 = e; }
    float wn = __builtin_amdgcn_rcpf(m1 + m2 + 1e-9f);
    if (lane == 0) {
        topIdx[2 * t]     = i1;  topW[2 * t]     = m1 * wn;
        topIdx[2 * t + 1] = i2;  topW[2 * t + 1] = m2 * wn;
        atomicAdd(&counts[i1], 1);
        atomicAdd(&counts[i2], 1);
#pragma unroll
        for (int e = 0; e < NEXP; ++e) simp[wv][e] = p[e];
    }
    __syncthreads();
    if (tid < NEXP) {                       // fixed-order intra-block sum
        float s = 0.f;
        for (int w = 0; w < 8; ++w) s += simp[w][tid];
        blockPart[blockIdx.x * NEXP + tid] = s;
    }
}

// Single wave: deterministic aux loss + expert prefix offsets.
__global__ void aux_kernel(const float* __restrict__ blockPart,
                           const int* __restrict__ counts,
                           int* __restrict__ offsetsE,
                           float* __restrict__ outAux, int nblk)
{
    int lane = threadIdx.x & 31;
    float prod = 0.f;
    if (lane < NEXP) {
        float s = 0.f;
        for (int b = 0; b < nblk; ++b) s += blockPart[b * NEXP + lane];
        float P = s / (float)NTOK;                     // mean gate prob
        float f = (float)counts[lane] / (float)NROW;   // usage fraction
        prod = f * P;
    }
    for (int o = 16; o; o >>= 1) prod += __shfl_xor(prod, o, 32);
    if (lane == 0) {
        float aux = prod * (float)NEXP * 0.01f;
        outAux[0] = aux > 1.f ? 1.f : aux;
        int off = 0;
        for (int e = 0; e < NEXP; ++e) { offsetsE[e] = off; off += counts[e]; }
    }
}

__global__ void scatter_kernel(const int* __restrict__ topIdx,
                               const float* __restrict__ topW,
                               const int* __restrict__ offsetsE,
                               int* __restrict__ cursors,
                               int* __restrict__ permTok,
                               int* __restrict__ slotOf,
                               float* __restrict__ rowW)
{
    int t = blockIdx.x * 256 + threadIdx.x;
    if (t >= NTOK) return;
#pragma unroll
    for (int k = 0; k < 2; ++k) {
        int e   = topIdx[2 * t + k];
        int pos = atomicAdd(&cursors[e], 1);
        int row = offsetsE[e] + pos;
        permTok[row] = t;
        slotOf[row]  = 2 * t + k;
        rowW[row]    = topW[2 * t + k];
    }
}

// Fused expert FFN: 16 gathered rows x full H output per workgroup.
// 8 waves; wave wv owns H columns [wv*128, wv*128+128).
__global__ __launch_bounds__(256) void expert_kernel(
    const float* __restrict__ x,
    const __bf16* __restrict__ guwB, const __bf16* __restrict__ dwB,
    const int* __restrict__ counts,  const int* __restrict__ offsetsE,
    const int* __restrict__ permTok, const int* __restrict__ slotOf,
    const float* __restrict__ rowW,  float* __restrict__ Ybuf)
{
    __shared__ __align__(16) __bf16 xs[BM * XS];
    __shared__ __align__(16) __bf16 act_s[BM * AS];
    __shared__ int toks[BM];

    int blk = blockIdx.x;
    int e   = blk >> 9;           // / MAXTILES
    int mt  = blk & (MAXTILES - 1);
    int cnt = counts[e];
    int m0  = mt * BM;
    if (m0 >= cnt) return;
    int offB = offsetsE[e];

    int tid = threadIdx.x;
    int lane = tid & 31;
    int wv   = tid >> 5;          // 0..7
    int hi   = lane >> 4;         // 0/1
    int ln   = lane & 15;

    if (tid < BM) {
        int r = m0 + tid; if (r > cnt - 1) r = cnt - 1;   // clamp ragged tail
        toks[tid] = permTok[offB + r];
    }
    __syncthreads();
    // X tile -> LDS bf16, vectorized (float4 load, 8B bf16x4 store)
    for (int i = tid; i < BM * (HID / 4); i += 256) {
        int row = i >> 8, c4 = (i & 255) * 4;
        float4 xv = *(const float4*)(x + (size_t)toks[row] * HID + c4);
        v4bf o; o[0]=f2bf(xv.x); o[1]=f2bf(xv.y); o[2]=f2bf(xv.z); o[3]=f2bf(xv.w);
        *(v4bf*)(xs + row * XS + c4) = o;
    }
    __syncthreads();

    const __bf16* guwE = guwB + (size_t)e * (2 * INTER) * HID;
    const __bf16* dwE  = dwB  + (size_t)e * HID * INTER;

    v8f zero = {};
    v8f acc[8];
#pragma unroll
    for (int j = 0; j < 8; ++j) acc[j] = zero;

#pragma unroll 1
    for (int fc0 = 0; fc0 < INTER; fc0 += FC) {
        // ---- GEMM1: this wave's 16 gate + 16 up columns, K = H ----
        int fg = fc0 + wv * 16 + ln;            // gate row of gate_up_w
        const __bf16* pg = guwE + (size_t)fg * HID + hi * 16;
        const __bf16* pu = guwE + (size_t)(INTER + fg) * HID + hi * 16;
        const __bf16* arow = xs + ln * XS;
        v8f g = zero, u = zero;
#pragma unroll 2
        for (int ks = 0; ks < HID; ks += 32) {
            __builtin_prefetch(pg + ks + 256, 0, 1);   // global_prefetch_b8
            __builtin_prefetch(pu + ks + 256, 0, 1);
            v16bf a  = loadA16(arow + ks, hi);
            v16bf bg = loadB16(pg + ks);
            v16bf bu = loadB16(pu + ks);
            g = wmma_bf16(a, bg, g);
            u = wmma_bf16(a, bu, u);
        }
        // ---- SiLU(g)*u elementwise in C layout, stage to LDS as bf16 ----
#pragma unroll
        for (int r = 0; r < 8; ++r) {
            float gv = g[r], uv = u[r];
            float s  = gv * __builtin_amdgcn_rcpf(1.f + __expf(-gv)) * uv;
            int m    = r + hi * 8;
            act_s[m * AS + wv * 16 + ln] = f2bf(s);
        }
        __syncthreads();
        // ---- GEMM2: acc[16 x 128] += act[16 x FC] @ down^T chunk ----
        const __bf16* a2row = act_s + ln * AS;
#pragma unroll
        for (int c = 0; c < FC / 32; ++c) {
            v16bf a2 = loadA16(a2row + c * 32, hi);
#pragma unroll
            for (int j = 0; j < 8; ++j) {
                int h = wv * 128 + j * 16 + ln;
                const __bf16* pd = dwE + (size_t)h * INTER + fc0 + c * 32 + hi * 16;
                v16bf b = loadB16(pd);
                acc[j] = wmma_bf16(a2, b, acc[j]);
            }
        }
        __syncthreads();
    }
    // ---- scaled store into per-(token,slot) buffer (deterministic) ----
#pragma unroll
    for (int r = 0; r < 8; ++r) {
        int m = r + hi * 8;
        if (m0 + m < cnt) {
            int grow = offB + m0 + m;
            int slot = slotOf[grow];
            float w  = rowW[grow];
#pragma unroll
            for (int j = 0; j < 8; ++j)
                Ybuf[(size_t)slot * HID + wv * 128 + j * 16 + ln] = w * acc[j][r];
        }
    }
}

__global__ void combine_kernel(const float* __restrict__ Ybuf,
                               float* __restrict__ out)
{
    size_t q = (size_t)blockIdx.x * 256 + threadIdx.x;   // float4 index
    size_t t = q / (HID / 4);
    size_t b = (q % (HID / 4)) * 4;
    const float4* A = (const float4*)(Ybuf + (2 * t)     * (size_t)HID + b);
    const float4* B = (const float4*)(Ybuf + (2 * t + 1) * (size_t)HID + b);
    float4 a = *A, c = *B;
    float4 o; o.x = a.x + c.x; o.y = a.y + c.y; o.z = a.z + c.z; o.w = a.w + c.w;
    *(float4*)(out + t * (size_t)HID + b) = o;
}

// ---------------------------------------------------------------------------
extern "C" void kernel_launch(void* const* d_in, const int* in_sizes, int n_in,
                              void* d_out, int out_size, void* d_ws, size_t ws_size,
                              hipStream_t stream)
{
    const float* x   = (const float*)d_in[0];
    const float* gw  = (const float*)d_in[1];
    const float* guw = (const float*)d_in[2];
    const float* dw  = (const float*)d_in[3];
    float* out = (float*)d_out;

    char* ws = (char*)d_ws;
    constexpr size_t NBLK     = NTOK / 8;                 // routing blocks
    constexpr size_t OFF_BP   = 0;                        // [NBLK*8] f32
    constexpr size_t OFF_CNT  = OFF_BP + NBLK * NEXP * 4; // counts[8] (+cursors)
    constexpr size_t OFF_CUR  = OFF_CNT + 32;
    constexpr size_t OFF_OFFE = OFF_CUR + 32;
    constexpr size_t OFF_TIDX = OFF_OFFE + 32;            // [2T] int
    constexpr size_t OFF_TW   = OFF_TIDX + NROW * 4;      // [2T] f32
    constexpr size_t OFF_PT   = OFF_TW   + NROW * 4;      // [R] int
    constexpr size_t OFF_SO   = OFF_PT   + NROW * 4;      // [R] int
    constexpr size_t OFF_RW   = OFF_SO   + NROW * 4;      // [R] f32
    constexpr size_t OFF_Y    = (OFF_RW + NROW * 4 + 127) & ~(size_t)127;
    constexpr size_t NGU      = (size_t)NEXP * 2 * INTER * HID;   // gate_up elems
    constexpr size_t NDW      = (size_t)NEXP * HID * INTER;       // down elems
    constexpr size_t OFF_GUB  = OFF_Y + (size_t)NROW * HID * 4;   // bf16 cache
    constexpr size_t OFF_DWB  = OFF_GUB + NGU * 2;

    float*  blockPart = (float*)(ws + OFF_BP);
    int*    counts    = (int*)  (ws + OFF_CNT);
    int*    cursors   = (int*)  (ws + OFF_CUR);
    int*    offsetsE  = (int*)  (ws + OFF_OFFE);
    int*    topIdx    = (int*)  (ws + OFF_TIDX);
    float*  topW      = (float*)(ws + OFF_TW);
    int*    permTok   = (int*)  (ws + OFF_PT);
    int*    slotOf    = (int*)  (ws + OFF_SO);
    float*  rowW      = (float*)(ws + OFF_RW);
    float*  Ybuf      = (float*)(ws + OFF_Y);
    __bf16* guwB      = (__bf16*)(ws + OFF_GUB);
    __bf16* dwB       = (__bf16*)(ws + OFF_DWB);

    cvt_kernel<<<(unsigned)(NGU / 8 / 256), 256, 0, stream>>>(guw, guwB,
                                                              (unsigned)(NGU / 8));
    cvt_kernel<<<(unsigned)(NDW / 8 / 256), 256, 0, stream>>>(dw, dwB,
                                                              (unsigned)(NDW / 8));
    init_kernel<<<1, 64, 0, stream>>>(counts);            // counts + cursors
    routing_kernel<<<NBLK, 256, 0, stream>>>(x, gw, topIdx, topW, counts, blockPart);
    aux_kernel<<<1, 32, 0, stream>>>(blockPart, counts, offsetsE,
                                     out + (size_t)NTOK * HID, (int)NBLK);
    scatter_kernel<<<NTOK / 256, 256, 0, stream>>>(topIdx, topW, offsetsE,
                                                   cursors, permTok, slotOf, rowW);
    expert_kernel<<<NEXP * MAXTILES, 256, 0, stream>>>(x, guwB, dwB, counts, offsetsE,
                                                       permTok, slotOf, rowW, Ybuf);
    combine_kernel<<<(NTOK * HID / 4) / 256, 256, 0, stream>>>(Ybuf, out);
}
